// LikelihoodRatio_42820823941556
// MI455X (gfx1250) — matrix operand order repaired and checked by reading, not compiled, not run on hardware
//
#include <hip/hip_runtime.h>
#include <cmath>

// ---------------------------------------------------------------------------
// Likelihood-ratio kernel for MI455X (gfx1250, wave32, WMMA).
//
// like[b,p] = sum_h pois(h; mu) * sum_e gauss(val; e, std) * binom[e,h]
// evaluated at val = x (observation) and val = mu (best fit).
//
// WMMA mapping (v_wmma_f32_16x16x32_f16):
//   A (16x32 f16): rows 0-7 = gauss(x) of 8 (b,p) rows, rows 8-15 = gauss(mu)
//                  of the same rows  -> both likelihoods in ONE accumulator set
//   B (32x16 f16): binom table, constant, pre-swizzled into fragment layout
//   C (16x16 f32): inner[16, h-tile], accumulated over 4 K-steps (e padded 128)
// C layout: lanes 0-15 hold the x-case rows, lanes 16-31 the mu-case rows,
// so ll_x - ll_mu is one shfl_xor(16).
//
// All exponentials use the RAW v_exp_f32 (base-2, flush-to-zero: underflowing
// probability terms are negligible by construction), log2(e) folded into
// hoisted constants / staged tables.
// ---------------------------------------------------------------------------

typedef __attribute__((ext_vector_type(16))) _Float16 v16h;
typedef __attribute__((ext_vector_type(8)))  float    v8f;

#define BATCH    2048
#define N_PMTS   253
#define N_PH     80          // Poisson photon domain
#define N_PE     120         // PE domain (padded to 128 for WMMA K)
#define N_PE_PAD 128
#define KSTEPS   4           // 4 * 32 = 128
#define NTILES   5           // 5 * 16 = 80 == N_PH
#define TOT_ROWS (BATCH * N_PMTS)          // 518144
#define ROWS_PER_WAVE 8                    // 8 (b,p) rows x {x, mu} = 16 M-rows
#define N_TILES (TOT_ROWS / ROWS_PER_WAVE) // 64768 (exact)
#define WAVES_PER_BLOCK 8
#define NUM_BLOCKS (N_TILES / WAVES_PER_BLOCK)  // 8096 (exact)

#define LOG2E 1.4426950408889634f
#define LN2   0.6931471805599453f

// Swizzled binom table: [KSTEPS][NTILES][32 lanes][16 halves] = 10240 f16 = 20480 B
#define BSWZ_ELEMS (KSTEPS * NTILES * 32 * 16)
#define LGT_OFF_BYTES 20480                // lgamma(h+1) table follows in d_ws

#if defined(__has_builtin) && __has_builtin(__builtin_amdgcn_rsqf)
#define RSQ(x) __builtin_amdgcn_rsqf(x)
#else
#define RSQ(x) rsqrtf(x)
#endif

// Raw hardware exp2/log2 (no denormal guard sequence).
#if defined(__has_builtin) && __has_builtin(__builtin_amdgcn_exp2f)
#define EXP2(x) __builtin_amdgcn_exp2f(x)
#else
#define EXP2(x) exp2f(x)
#endif
#if defined(__has_builtin) && __has_builtin(__builtin_amdgcn_logf)
#define LOG2(x) __builtin_amdgcn_logf(x)
#else
#define LOG2(x) log2f(x)
#endif

// ---------------------------------------------------------------------------
// Init kernel 1: zero output accumulator and build lgamma(n+1) = log(n!) table
// via a log-sum (exact for integers; avoids inlining the huge lgammaf).
// ---------------------------------------------------------------------------
__global__ void lr_table_kernel(float* __restrict__ lgt,
                                float* __restrict__ out) {
    int t = blockIdx.x * blockDim.x + threadIdx.x;
    if (t < BATCH) out[t] = 0.0f;            // we own zeroing of d_out
    if (t < N_PH) {
        float s = 0.0f;
        for (int i = 2; i <= t; ++i) s += __logf((float)i);
        lgt[t] = s;                          // lgamma(t+1), natural log
    }
}

// ---------------------------------------------------------------------------
// Init kernel 2: binom table in the per-lane WMMA B-fragment layout:
//   fragment (ks, nt): lane L holds column h = nt*16 + (L&15),
//   halves j=0..15 hold e = ks*32 + ((L>=16)?16:0) + j
// binom pmf over k = e - h requires 0 <= k <= h and e < 120.
// ---------------------------------------------------------------------------
__global__ void lr_binom_kernel(const float* __restrict__ lgt,
                                _Float16* __restrict__ bswz) {
    int t = blockIdx.x * blockDim.x + threadIdx.x;
    if (t >= KSTEPS * NTILES * 32) return;   // 640 builder threads
    int lane = t & 31;
    int frag = t >> 5;                       // ks*NTILES + nt
    int nt   = frag % NTILES;
    int ks   = frag / NTILES;

    int h     = nt * 16 + (lane & 15);              // n_ph (0..79)
    int kbase = ks * 32 + ((lane >> 4) << 4);       // e base for this lane half

    const float logp   = __logf(0.2f);
    const float log1mp = __logf(0.8f);
    float lg_h1 = lgt[h];

    for (int j = 0; j < 16; ++j) {
        int e = kbase + j;                          // n_pe (0..127)
        int k = e - h;
        float v = 0.0f;
        if (e < N_PE && k >= 0 && k <= h) {
            float lpmf = lg_h1 - lgt[k] - lgt[h - k]
                       + (float)k * logp + (float)(h - k) * log1mp;
            v = __expf(lpmf);
        }
        bswz[(size_t)t * 16 + j] = (_Float16)v;
    }
}

// ---------------------------------------------------------------------------
// Main kernel: one wave per 8-row tile; x-case and mu-case share one WMMA tile.
// ---------------------------------------------------------------------------
__global__ __launch_bounds__(WAVES_PER_BLOCK * 32, 1)
void lr_main_kernel(const float* __restrict__ pred,
                    const float* __restrict__ observed,
                    const float* __restrict__ stdv,
                    const _Float16* __restrict__ bswz,
                    const float* __restrict__ lgt,
                    float* __restrict__ out) {
    __shared__ __align__(32) _Float16 ldsB[BSWZ_ELEMS];     // 20480 B
    __shared__ __align__(16) float2 lds_g[N_PE_PAD]; // {rsq(max(e,1)), e*rsq}
    __shared__ float lds_lgh2[N_PH];                 // lgamma(h+1) * log2(e)
    __shared__ float lds_mu2[WAVES_PER_BLOCK][ROWS_PER_WAVE];   // mu * log2(e)
    __shared__ float lds_lmu2[WAVES_PER_BLOCK][ROWS_PER_WAVE];  // log2(mu+eps)
    __shared__ float lds_nan[WAVES_PER_BLOCK][ROWS_PER_WAVE];

    // Cooperative stage of binom table + small constant tables into LDS.
    {
        const uint32_t* src = (const uint32_t*)bswz;
        uint32_t* dst = (uint32_t*)ldsB;
        for (int i = threadIdx.x; i < BSWZ_ELEMS / 2; i += blockDim.x)
            dst[i] = src[i];
        for (int i = threadIdx.x; i < N_PH; i += blockDim.x)
            lds_lgh2[i] = lgt[i] * LOG2E;
        for (int i = threadIdx.x; i < N_PE_PAD; i += blockDim.x) {
            float ef = (float)i;
            float g  = RSQ(fmaxf(ef, 1.0f));
            lds_g[i] = make_float2(g, ef * g);
        }
    }

    const int wave = threadIdx.x >> 5;
    const int lane = threadIdx.x & 31;
    const int rowM = lane & 15;          // WMMA A row: 0-7 -> x-case, 8-15 -> mu-case
    const int hi   = lane >> 4;          // 0: K-low half, 1: K-high half
    const int prow = rowM & 7;           // physical (b,p) row within the tile
    const int tile = blockIdx.x * WAVES_PER_BLOCK + wave;   // < N_TILES exactly

    const int r = tile * ROWS_PER_WAVE + prow;   // flattened (b,p) row
    const int b = r / N_PMTS;
    const int p = r - b * N_PMTS;

    float mu  = fmaxf(pred[(size_t)r], 1e-10f);
    float obs = observed[(size_t)r];
    bool  bad = (obs != obs);                    // NaN-safe mask
    float x   = bad ? 0.0f : obs;
    float sd  = stdv[p];
    float val = (rowM < 8) ? x : mu;             // which likelihood this A-row feeds
    float inv_sd = 1.0f / sd;
    float cneg2 = -0.5f * LOG2E * inv_sd * inv_sd;  // gauss exponent, base-2
    float nrm   = inv_sd * 0.3989422804014327f;     // 1/(sd*sqrt(2*pi))

    if (lane < ROWS_PER_WAVE) {                  // per-row stats for Poisson stage
        lds_mu2[wave][lane]  = mu * LOG2E;
        lds_lmu2[wave][lane] = LOG2(mu + 1e-10f);   // log2(mu+eps)
        lds_nan[wave][lane]  = bad ? 1.0f : 0.0f;
    }
    __syncthreads();

    // inner[16 M-rows, 80 h] accumulators (x-case rows 0-7, mu-case rows 8-15)
    v8f c[NTILES];
#pragma unroll
    for (int t = 0; t < NTILES; ++t)
        c[t] = (v8f){0.f,0.f,0.f,0.f,0.f,0.f,0.f,0.f};

#pragma unroll
    for (int ks = 0; ks < KSTEPS; ++ks) {
        // A fragment: e runs in two contiguous octets per lane:
        //   j=0..7  -> e = ks*32 + hi*16 + j
        //   j=8..15 -> e = ks*32 + hi*16 + 8 + (j-8)
        const int e0 = ks * 32 + (hi << 4);
        const int e1 = e0 + 8;

        // Batch the 16 table entries into registers first (DS loads clause,
        // single wait), then run the 16 exp chains breadth-first.
        float2 gbuf[16];
        {
            const float4* gp = (const float4*)lds_g;
            float4 q0 = gp[(e0 >> 1) + 0], q1 = gp[(e0 >> 1) + 1];
            float4 q2 = gp[(e0 >> 1) + 2], q3 = gp[(e0 >> 1) + 3];
            float4 q4 = gp[(e1 >> 1) + 0], q5 = gp[(e1 >> 1) + 1];
            float4 q6 = gp[(e1 >> 1) + 2], q7 = gp[(e1 >> 1) + 3];
            gbuf[0]  = make_float2(q0.x, q0.y); gbuf[1]  = make_float2(q0.z, q0.w);
            gbuf[2]  = make_float2(q1.x, q1.y); gbuf[3]  = make_float2(q1.z, q1.w);
            gbuf[4]  = make_float2(q2.x, q2.y); gbuf[5]  = make_float2(q2.z, q2.w);
            gbuf[6]  = make_float2(q3.x, q3.y); gbuf[7]  = make_float2(q3.z, q3.w);
            gbuf[8]  = make_float2(q4.x, q4.y); gbuf[9]  = make_float2(q4.z, q4.w);
            gbuf[10] = make_float2(q5.x, q5.y); gbuf[11] = make_float2(q5.z, q5.w);
            gbuf[12] = make_float2(q6.x, q6.y); gbuf[13] = make_float2(q6.z, q6.w);
            gbuf[14] = make_float2(q7.x, q7.y); gbuf[15] = make_float2(q7.z, q7.w);
        }

        float tz[16], ex[16];
#pragma unroll
        for (int j = 0; j < 16; ++j)
            tz[j] = fmaf(val, gbuf[j].x, -gbuf[j].y);     // (val - e)/sqrt(e)
#pragma unroll
        for (int j = 0; j < 16; ++j)
            ex[j] = EXP2(cneg2 * tz[j] * tz[j]);          // raw v_exp_f32
        v16h a;
#pragma unroll
        for (int j = 0; j < 16; ++j)
            a[j] = (_Float16)(ex[j] * (gbuf[j].x * nrm));

#pragma unroll
        for (int t = 0; t < NTILES; ++t) {
            v16h bf = *(const v16h*)&ldsB[((ks * NTILES + t) * 32 + lane) * 16];
            c[t] = __builtin_amdgcn_wmma_f32_16x16x32_f16(
                       false, a, false, bf, (short)0, c[t], false, false);
        }
    }

    // Poisson weighting: C layout -> lane holds h = t*16 + (lane&15);
    // VGPR v holds M-row v + 8*hi: lo lanes = x-case rows, hi lanes = mu-case rows.
    float hhf[NTILES], lghv2[NTILES];
#pragma unroll
    for (int t = 0; t < NTILES; ++t) {
        int hh = t * 16 + rowM;
        hhf[t]   = (float)hh;
        lghv2[t] = lds_lgh2[hh];
    }

    float acc[ROWS_PER_WAVE];
#pragma unroll
    for (int v = 0; v < ROWS_PER_WAVE; ++v) {
        float m2  = lds_mu2[wave][v];
        float lm2 = lds_lmu2[wave][v];
        float s   = 0.0f;
#pragma unroll
        for (int t = 0; t < NTILES; ++t) {
            // pois(h; mu) = 2^(h*log2(mu) - mu*log2e - lgamma(h+1)*log2e)
            float w = EXP2(fmaf(hhf[t], lm2, -m2 - lghv2[t]));
            s = fmaf(c[t][v], w, s);
        }
        acc[v] = s;
    }

    // Butterfly-reduce over each 16-lane group (masks 1..8 stay in-group).
#pragma unroll
    for (int m = 1; m <= 8; m <<= 1)
#pragma unroll
        for (int v = 0; v < ROWS_PER_WAVE; ++v)
            acc[v] += __shfl_xor(acc[v], m, 32);

    // Compile-time select of this lane's row (avoids dynamic register indexing).
    float sel = acc[0];
#pragma unroll
    for (int v = 1; v < ROWS_PER_WAVE; ++v)
        if (prow == v) sel = acc[v];

    float lg = LOG2(sel + 1.17549435e-38f);      // log2(like + float32 tiny)
    float lg_other = __shfl_xor(lg, 16, 32);     // partner group's likelihood

    // Lanes 0-7 finalize: lg = log2 L(x), lg_other = log2 L(mu) for this row.
    if (lane < ROWS_PER_WAVE) {
        float ratio = (-2.0f * LN2) * (lg - lg_other);   // -2*(ln L_x - ln L_mu)
        if (lds_nan[wave][lane] != 0.0f) ratio = 100000.0f;
        int rr = tile * ROWS_PER_WAVE + lane;
        int bb = rr / N_PMTS;
        atomicAdd(&out[bb], ratio);
    }
}

// ---------------------------------------------------------------------------
extern "C" void kernel_launch(void* const* d_in, const int* in_sizes, int n_in,
                              void* d_out, int out_size, void* d_ws, size_t ws_size,
                              hipStream_t stream) {
    const float* pred     = (const float*)d_in[0];
    const float* observed = (const float*)d_in[1];
    const float* stdv     = (const float*)d_in[2];
    float*       out      = (float*)d_out;
    _Float16*    bswz     = (_Float16*)d_ws;
    float*       lgt      = (float*)((char*)d_ws + LGT_OFF_BYTES);

    // 1) log-factorial table + zero output accumulator.
    lr_table_kernel<<<8, 256, 0, stream>>>(lgt, out);
    // 2) Pre-swizzled binom B-matrix table.
    lr_binom_kernel<<<5, 128, 0, stream>>>(lgt, bswz);
    // 3) Main WMMA likelihood-ratio kernel: one wave per 8-row tile (x & mu fused).
    lr_main_kernel<<<NUM_BLOCKS, WAVES_PER_BLOCK * 32, 0, stream>>>(
        pred, observed, stdv, bswz, lgt, out);
}
